// TransformerEncoderLayer_43336220016868
// MI455X (gfx1250) — matrix operand order repaired and checked by reading, not compiled, not run on hardware
//
#include <hip/hip_runtime.h>

// ---------------------------------------------------------------------------
// Deformable-DETR transformer encoder layer for MI455X (gfx1250, wave32).
// GEMMs: v_wmma_f32_16x16x32_bf16, 32x32 wave tiles, double-buffered LDS fed
// by CDNA5 async-to-LDS DMA (global_load_async_to_lds_b128 + ASYNCcnt).
// ---------------------------------------------------------------------------

#define NB    8
#define LQDIM 5440
#define MTOT  (NB * LQDIM)   // 43520 (divisible by 128)
#define CDIM  256
#define NHEAD 8
#define NLVL  4
#define NPTS  4
#define DHEAD 32
#define DFFN  1024

typedef __bf16 bf16_t;
typedef __attribute__((ext_vector_type(16))) bf16_t v16bf;
typedef __attribute__((ext_vector_type(8)))  float  v8f;

static __device__ __forceinline__ bf16_t f2bf(float f) {
    union { float f; unsigned u; } x; x.f = f;
    unsigned r = x.u + 0x7FFFu + ((x.u >> 16) & 1u);   // round-to-nearest-even
    union { unsigned short s; bf16_t b; } y; y.s = (unsigned short)(r >> 16);
    return y.b;
}

static __device__ __forceinline__ unsigned lds_off(const void* p) {
    // Flat addresses of LDS objects carry the LDS byte offset in addr[31:0].
    return (unsigned)(unsigned long long)p;
}

// ---------------------------------------------------------------------------
// Weight transpose + f32 -> bf16:  in[K, Ncols] row-major  ->  out[Ncols, K]
// ---------------------------------------------------------------------------
__global__ __launch_bounds__(256) void wtrans_kernel(
    const float* __restrict__ in, bf16_t* __restrict__ out, int K, int Ncols)
{
    int i = blockIdx.x * 256 + threadIdx.x;
    if (i >= K * Ncols) return;
    int k = i / Ncols, n = i % Ncols;
    out[(size_t)n * K + k] = f2bf(in[i]);
}

// ---------------------------------------------------------------------------
// query = src + pos ; bf16 copies of src and query for the WMMA GEMMs
// ---------------------------------------------------------------------------
__global__ __launch_bounds__(256) void prep_kernel(
    const float* __restrict__ src, const float* __restrict__ pos,
    bf16_t* __restrict__ src_bf, bf16_t* __restrict__ q_bf)
{
    size_t i = (size_t)blockIdx.x * 256 + threadIdx.x;
    float s = src[i];
    src_bf[i] = f2bf(s);
    q_bf[i]   = f2bf(s + pos[i]);
}

// ---------------------------------------------------------------------------
// Generic bf16 WMMA GEMM:  C[M,Ncols] = A[M,K] * Bt[Ncols,K]^T + bias
// Block = 256 threads = 8 waves; block tile 128x64; wave tile 32x32
// (four v_wmma_f32_16x16x32_bf16 per 32-wide K step).
// LDS double-buffered, filled with global_load_async_to_lds_b128; the DMA for
// tile kt+1 overlaps the WMMAs of tile kt (s_wait_asynccnt 3 = prev tile done,
// since async loads complete in order per wave).
// Epilogue: optional ReLU, optional per-row zero mask, f32 and/or bf16 store.
// ---------------------------------------------------------------------------
__global__ __launch_bounds__(256) void gemm_bf16_kernel(
    const bf16_t* __restrict__ A, const bf16_t* __restrict__ Bt,
    const float* __restrict__ bias, float* __restrict__ Cf,
    bf16_t* __restrict__ Cb, const unsigned char* __restrict__ rowmask,
    int Ncols, int K, int relu)
{
    __shared__ bf16_t As[2][128][34];   // pitch 34 (68 B) -> conflict-free gathers
    __shared__ bf16_t Bs[2][64][34];

    const int t       = threadIdx.x;
    const int lane    = t & 31;
    const int wave    = t >> 5;
    const int half    = lane >> 4;
    const int l16     = lane & 15;
    const int waveRow = wave >> 1;     // 0..3  (32 rows each)
    const int waveCol = wave & 1;      // 0..1  (32 cols each)
    const int blockN  = blockIdx.x * 64;
    const int blockM  = blockIdx.y * 128;
    const int lr      = t >> 2;        // 0..63 : tile row for cooperative load
    const int lseg    = t & 3;         // 0..3  : 8-element K segment

    const int gcol0 = blockN + waveCol * 32 + l16;
    const int gcol1 = gcol0 + 16;
    const float b0 = bias ? bias[gcol0] : 0.0f;
    const float b1 = bias ? bias[gcol1] : 0.0f;
    v8f acc00, acc01, acc10, acc11;
#pragma unroll
    for (int i = 0; i < 8; ++i) {
        acc00[i] = b0; acc01[i] = b1; acc10[i] = b0; acc11[i] = b1;
    }

    const int ksteps = K >> 5;

    // Issue the async DMA for one K tile into LDS buffer `buf`:
    // per thread: A rows lr and lr+64, B row lr (16 B each) -> 3 async B128s.
    auto issue_tile = [&](int kt, int buf) {
        const size_t kof = (size_t)kt * 32 + lseg * 8;
        const unsigned la0 = lds_off(&As[buf][lr][lseg * 8]);
        const unsigned la1 = lds_off(&As[buf][lr + 64][lseg * 8]);
        const unsigned lb  = lds_off(&Bs[buf][lr][lseg * 8]);
        const unsigned long long ga0 =
            (unsigned long long)(A + (size_t)(blockM + lr) * K + kof);
        const unsigned long long ga1 =
            (unsigned long long)(A + (size_t)(blockM + lr + 64) * K + kof);
        const unsigned long long gb =
            (unsigned long long)(Bt + (size_t)(blockN + lr) * K + kof);
        asm volatile(
            "global_load_async_to_lds_b128 %0, %3, off\n\t"
            "global_load_async_to_lds_b128 %1, %4, off\n\t"
            "global_load_async_to_lds_b128 %2, %5, off"
            :: "v"(la0), "v"(la1), "v"(lb), "v"(ga0), "v"(ga1), "v"(gb)
            : "memory");
    };

    issue_tile(0, 0);

    for (int kt = 0; kt < ksteps; ++kt) {
        const int buf = kt & 1;
        if (kt + 1 < ksteps) {
            issue_tile(kt + 1, buf ^ 1);                    // overlaps compute
            asm volatile("s_wait_asynccnt 0x3" ::: "memory"); // tile kt landed
        } else {
            asm volatile("s_wait_asynccnt 0x0" ::: "memory");
        }
        __syncthreads();

        // --- build fragments per ISA lane layouts (CDNA5 ISA §7.12.2) ---
        union { v16bf v; unsigned u[8]; } a0, a1, fb0, fb1;
        const bf16_t* ap0 = &As[buf][waveRow * 32 + l16][0];
        const bf16_t* ap1 = ap0 + 16 * 34;
#pragma unroll
        for (int v = 0; v < 4; ++v) {
            // A 16x32: lanes 0-15: K0-7 then K16-23 ; lanes 16-31: K8-15, K24-31
            a0.u[v]     = *(const unsigned*)(ap0 + half * 8 + 2 * v);
            a0.u[v + 4] = *(const unsigned*)(ap0 + 16 + half * 8 + 2 * v);
            a1.u[v]     = *(const unsigned*)(ap1 + half * 8 + 2 * v);
            a1.u[v + 4] = *(const unsigned*)(ap1 + 16 + half * 8 + 2 * v);
        }
        const bf16_t* bp0 = &Bs[buf][waveCol * 32 + l16][0];
        const bf16_t* bp1 = bp0 + 16 * 34;
#pragma unroll
        for (int v = 0; v < 8; ++v) {
            // B 32x16: lanes 0-15: K0-7 then K16-23 ; lanes 16-31: K8-15, K24-31
            fb0.u[v] = *(const unsigned*)(bp0 + half * 8 + 2 * v + (v >> 2) * 8);
            fb1.u[v] = *(const unsigned*)(bp1 + half * 8 + 2 * v + (v >> 2) * 8);
        }

        acc00 = __builtin_amdgcn_wmma_f32_16x16x32_bf16(
            false, a0.v, false, fb0.v, (short)0, acc00, false, false);
        acc01 = __builtin_amdgcn_wmma_f32_16x16x32_bf16(
            false, a0.v, false, fb1.v, (short)0, acc01, false, false);
        acc10 = __builtin_amdgcn_wmma_f32_16x16x32_bf16(
            false, a1.v, false, fb0.v, (short)0, acc10, false, false);
        acc11 = __builtin_amdgcn_wmma_f32_16x16x32_bf16(
            false, a1.v, false, fb1.v, (short)0, acc11, false, false);
        __syncthreads();
    }

    // --- epilogue: C/D layout -> row = r + 8*half, col = lane&15 ---
#pragma unroll
    for (int r = 0; r < 8; ++r) {
        const int grow0 = blockM + waveRow * 32 + r + 8 * half;
        const int grow1 = grow0 + 16;
        float v00 = acc00[r], v01 = acc01[r], v10 = acc10[r], v11 = acc11[r];
        if (relu) {
            v00 = fmaxf(v00, 0.0f); v01 = fmaxf(v01, 0.0f);
            v10 = fmaxf(v10, 0.0f); v11 = fmaxf(v11, 0.0f);
        }
        if (rowmask) {
            if (rowmask[grow0]) { v00 = 0.0f; v01 = 0.0f; }
            if (rowmask[grow1]) { v10 = 0.0f; v11 = 0.0f; }
        }
        if (Cf) {
            Cf[(size_t)grow0 * Ncols + gcol0] = v00;
            Cf[(size_t)grow0 * Ncols + gcol1] = v01;
            Cf[(size_t)grow1 * Ncols + gcol0] = v10;
            Cf[(size_t)grow1 * Ncols + gcol1] = v11;
        }
        if (Cb) {
            Cb[(size_t)grow0 * Ncols + gcol0] = f2bf(v00);
            Cb[(size_t)grow0 * Ncols + gcol1] = f2bf(v01);
            Cb[(size_t)grow1 * Ncols + gcol0] = f2bf(v10);
            Cb[(size_t)grow1 * Ncols + gcol1] = f2bf(v11);
        }
    }
}

// ---------------------------------------------------------------------------
// Softmax over the 16 (level,point) attention logits per (token, head).
// ---------------------------------------------------------------------------
__global__ __launch_bounds__(256) void softmax16_kernel(float* __restrict__ logits)
{
    const int i = blockIdx.x * 256 + threadIdx.x;      // one (row, head)
    float* p = logits + (size_t)i * 16;
    float v[16];
    float m = -1e30f;
#pragma unroll
    for (int j = 0; j < 16; ++j) { v[j] = p[j]; m = fmaxf(m, v[j]); }
    float s = 0.0f;
#pragma unroll
    for (int j = 0; j < 16; ++j) { v[j] = __expf(v[j] - m); s += v[j]; }
    const float inv = 1.0f / s;
#pragma unroll
    for (int j = 0; j < 16; ++j) p[j] = v[j] * inv;
}

// ---------------------------------------------------------------------------
// Deformable attention sampling. One wave per (token,head); lane = head dim.
// Each corner gather is one coalesced 128 B load; value (44 MB) lives in L2.
// ---------------------------------------------------------------------------
__global__ __launch_bounds__(256) void msda_sample_kernel(
    const float* __restrict__ value, const float* __restrict__ off,
    const float* __restrict__ aw, const float* __restrict__ ref,
    const int* __restrict__ shapes, const int* __restrict__ lstart,
    float* __restrict__ outf, bf16_t* __restrict__ outb)
{
    const int lane = threadIdx.x & 31;                 // d in [0,32)
    const int gid  = blockIdx.x * 8 + (threadIdx.x >> 5);
    const int row  = gid >> 3;                         // n*LQ + q
    const int h    = gid & 7;
    const int n    = row / LQDIM;

    const float* offp = off + (size_t)row * 256 + h * 32;  // (l*NP+p)*2 + {x,y}
    const float* awp  = aw  + (size_t)row * 128 + h * 16;
    const float* refp = ref + (size_t)row * 8;             // [NL,2]

    float acc = 0.0f;
#pragma unroll
    for (int l = 0; l < NLVL; ++l) {
        const int H = shapes[2 * l], W = shapes[2 * l + 1];
        const float* vbase =
            value + ((size_t)(n * LQDIM + lstart[l])) * 256 + h * 32 + lane;
        const float rx = refp[2 * l], ry = refp[2 * l + 1];
#pragma unroll
        for (int p = 0; p < NPTS; ++p) {
            const float ox = offp[(l * 4 + p) * 2];
            const float oy = offp[(l * 4 + p) * 2 + 1];
            const float w  = awp[l * 4 + p];
            // (gx+1)*W/2 - 0.5 with gx = 2*(rx + ox/W) - 1  ==  rx*W + ox - 0.5
            const float x = rx * (float)W + ox - 0.5f;
            const float y = ry * (float)H + oy - 0.5f;
            const float xf = floorf(x), yf = floorf(y);
            const float wx = x - xf, wy = y - yf;
            const int x0 = (int)xf, y0 = (int)yf;
            float s = 0.0f;
            if (x0 >= 0 && x0 < W && y0 >= 0 && y0 < H)
                s += vbase[(size_t)(y0 * W + x0) * 256] * (1.f - wx) * (1.f - wy);
            if (x0 + 1 >= 0 && x0 + 1 < W && y0 >= 0 && y0 < H)
                s += vbase[(size_t)(y0 * W + x0 + 1) * 256] * wx * (1.f - wy);
            if (x0 >= 0 && x0 < W && y0 + 1 >= 0 && y0 + 1 < H)
                s += vbase[(size_t)((y0 + 1) * W + x0) * 256] * (1.f - wx) * wy;
            if (x0 + 1 >= 0 && x0 + 1 < W && y0 + 1 >= 0 && y0 + 1 < H)
                s += vbase[(size_t)((y0 + 1) * W + x0 + 1) * 256] * wx * wy;
            acc += w * s;
        }
    }
    outf[(size_t)row * 256 + h * 32 + lane] = acc;
    outb[(size_t)row * 256 + h * 32 + lane] = f2bf(acc);
}

// ---------------------------------------------------------------------------
// out = LayerNorm(a + b) * g + be. One wave per row (8 f32 per lane),
// wave32 __shfl_xor tree reduction. Optional bf16 copy for the next GEMM.
// ---------------------------------------------------------------------------
__global__ __launch_bounds__(256) void add_ln_kernel(
    const float* __restrict__ a, const float* __restrict__ b,
    const float* __restrict__ g, const float* __restrict__ be,
    float* __restrict__ outf, bf16_t* __restrict__ outb)
{
    const int lane = threadIdx.x & 31;
    const int row  = blockIdx.x * 8 + (threadIdx.x >> 5);
    const float* pa = a + (size_t)row * 256 + lane * 8;
    const float* pb = b + (size_t)row * 256 + lane * 8;
    float v[8], s = 0.0f, ss = 0.0f;
#pragma unroll
    for (int i = 0; i < 8; ++i) { v[i] = pa[i] + pb[i]; s += v[i]; ss += v[i] * v[i]; }
#pragma unroll
    for (int o = 16; o > 0; o >>= 1) {
        s  += __shfl_xor(s,  o, 32);
        ss += __shfl_xor(ss, o, 32);
    }
    const float mean = s * (1.0f / 256.0f);
    const float var  = ss * (1.0f / 256.0f) - mean * mean;
    const float rstd = rsqrtf(var + 1e-5f);
    float*  po  = outf + (size_t)row * 256 + lane * 8;
    bf16_t* pob = outb ? outb + (size_t)row * 256 + lane * 8 : (bf16_t*)0;
#pragma unroll
    for (int i = 0; i < 8; ++i) {
        const int c = lane * 8 + i;
        const float o = (v[i] - mean) * rstd * g[c] + be[c];
        po[i] = o;
        if (pob) pob[i] = f2bf(o);
    }
}

// ---------------------------------------------------------------------------
extern "C" void kernel_launch(void* const* d_in, const int* in_sizes, int n_in,
                              void* d_out, int out_size, void* d_ws, size_t ws_size,
                              hipStream_t stream)
{
    (void)in_sizes; (void)n_in; (void)out_size; (void)ws_size;
    const float* src    = (const float*)d_in[0];
    const float* pos    = (const float*)d_in[1];
    const float* ref    = (const float*)d_in[2];
    const float* w_off  = (const float*)d_in[3];
    const float* b_off  = (const float*)d_in[4];
    const float* w_attn = (const float*)d_in[5];
    const float* b_attn = (const float*)d_in[6];
    const float* w_val  = (const float*)d_in[7];
    const float* b_val  = (const float*)d_in[8];
    const float* w_out  = (const float*)d_in[9];
    const float* b_out  = (const float*)d_in[10];
    const float* g1     = (const float*)d_in[11];
    const float* be1    = (const float*)d_in[12];
    const float* w_ff1  = (const float*)d_in[13];
    const float* b_ff1  = (const float*)d_in[14];
    const float* w_ff2  = (const float*)d_in[15];
    const float* b_ff2  = (const float*)d_in[16];
    const float* g2     = (const float*)d_in[17];
    const float* be2    = (const float*)d_in[18];
    const unsigned char* pmask = (const unsigned char*)d_in[19];
    const int* shapes   = (const int*)d_in[20];
    const int* lstart   = (const int*)d_in[21];
    float* out          = (float*)d_out;

    // ------- workspace carve-out (all buffers fully rewritten each call) ----
    char* ws = (char*)d_ws;
    size_t o = 0;
    auto carve = [&](size_t bytes) -> char* {
        char* p = ws + o;
        o += (bytes + 255) & ~((size_t)255);
        return p;
    };
    bf16_t* wv_t  = (bf16_t*)carve((size_t)CDIM * CDIM * 2);   // w_val^T
    bf16_t* wo_t  = (bf16_t*)carve((size_t)CDIM * 256  * 2);   // w_off^T
    bf16_t* wa_t  = (bf16_t*)carve((size_t)CDIM * 128  * 2);   // w_attn^T
    bf16_t* wu_t  = (bf16_t*)carve((size_t)CDIM * CDIM * 2);   // w_out^T
    bf16_t* wf1_t = (bf16_t*)carve((size_t)CDIM * DFFN * 2);   // w_ff1^T
    bf16_t* wf2_t = (bf16_t*)carve((size_t)DFFN * CDIM * 2);   // w_ff2^T
    bf16_t* src_bf = (bf16_t*)carve((size_t)MTOT * CDIM * 2);
    bf16_t* q_bf   = (bf16_t*)carve((size_t)MTOT * CDIM * 2);
    float*  valf   = (float*)carve((size_t)MTOT * CDIM * 4);
    float*  offv   = (float*)carve((size_t)MTOT * 256 * 4);
    float*  attn   = (float*)carve((size_t)MTOT * 128 * 4);    // logits -> softmax in place
    float*  sampf  = (float*)carve((size_t)MTOT * CDIM * 4);
    bf16_t* sampb  = (bf16_t*)carve((size_t)MTOT * CDIM * 2);
    float*  src2   = (float*)carve((size_t)MTOT * CDIM * 4);
    float*  xf     = (float*)carve((size_t)MTOT * CDIM * 4);
    bf16_t* xb     = (bf16_t*)carve((size_t)MTOT * CDIM * 2);
    bf16_t* hb     = (bf16_t*)carve((size_t)MTOT * DFFN * 2);
    float*  yf     = (float*)carve((size_t)MTOT * CDIM * 4);

    // ------- weight conversion (tiny) --------------------------------------
    wtrans_kernel<<<(CDIM * CDIM) / 256, 256, 0, stream>>>(w_val,  wv_t,  CDIM, CDIM);
    wtrans_kernel<<<(CDIM * 256 ) / 256, 256, 0, stream>>>(w_off,  wo_t,  CDIM, 256);
    wtrans_kernel<<<(CDIM * 128 ) / 256, 256, 0, stream>>>(w_attn, wa_t,  CDIM, 128);
    wtrans_kernel<<<(CDIM * CDIM) / 256, 256, 0, stream>>>(w_out,  wu_t,  CDIM, CDIM);
    wtrans_kernel<<<(CDIM * DFFN) / 256, 256, 0, stream>>>(w_ff1,  wf1_t, CDIM, DFFN);
    wtrans_kernel<<<(DFFN * CDIM) / 256, 256, 0, stream>>>(w_ff2,  wf2_t, DFFN, CDIM);

    // ------- activations ----------------------------------------------------
    prep_kernel<<<MTOT, 256, 0, stream>>>(src, pos, src_bf, q_bf);

    const dim3 g256(CDIM / 64, MTOT / 128);    // (4, 340)
    const dim3 g128(128 / 64,  MTOT / 128);    // (2, 340)
    const dim3 g1024(DFFN / 64, MTOT / 128);   // (16, 340)

    // value = mask0(src @ w_val + b_val)
    gemm_bf16_kernel<<<g256, 256, 0, stream>>>(src_bf, wv_t, b_val, valf,
                                               (bf16_t*)0, pmask, CDIM, CDIM, 0);
    // sampling offsets = query @ w_off + b_off
    gemm_bf16_kernel<<<g256, 256, 0, stream>>>(q_bf, wo_t, b_off, offv,
                                               (bf16_t*)0, (const unsigned char*)0, 256, CDIM, 0);
    // attention logits = query @ w_attn + b_attn, then softmax over 16
    gemm_bf16_kernel<<<g128, 256, 0, stream>>>(q_bf, wa_t, b_attn, attn,
                                               (bf16_t*)0, (const unsigned char*)0, 128, CDIM, 0);
    softmax16_kernel<<<(MTOT * NHEAD) / 256, 256, 0, stream>>>(attn);

    // deformable bilinear sampling
    msda_sample_kernel<<<MTOT, 256, 0, stream>>>(valf, offv, attn, ref,
                                                 shapes, lstart, sampf, sampb);

    // src2 = samp @ w_out + b_out ; x = LN(src + src2)
    gemm_bf16_kernel<<<g256, 256, 0, stream>>>(sampb, wu_t, b_out, src2,
                                               (bf16_t*)0, (const unsigned char*)0, CDIM, CDIM, 0);
    add_ln_kernel<<<MTOT / 8, 256, 0, stream>>>(src, src2, g1, be1, xf, xb);

    // h = relu(x @ w_ff1 + b_ff1)  (bf16 only) ; y = h @ w_ff2 + b_ff2
    gemm_bf16_kernel<<<g1024, 256, 0, stream>>>(xb, wf1_t, b_ff1, (float*)0,
                                                hb, (const unsigned char*)0, DFFN, CDIM, 1);
    gemm_bf16_kernel<<<g256, 256, 0, stream>>>(hb, wf2_t, b_ff2, yf,
                                               (bf16_t*)0, (const unsigned char*)0, CDIM, DFFN, 0);

    // out = LN(x + y)
    add_ln_kernel<<<MTOT / 8, 256, 0, stream>>>(xf, yf, g2, be2, out, (bf16_t*)0);
}
